// mpnn_22771916604017
// MI455X (gfx1250) — matrix operand order repaired and checked by reading, not compiled, not run on hardware
//
#include <hip/hip_runtime.h>
#include <hip/hip_bf16.h>
#include <stdint.h>

// ---------------------------------------------------------------------------
// MI455X (gfx1250) implementation of the MPNN reference.
// All dense linears run through v_wmma_f32_16x16x32_bf16 (wave32).
// Main GEMM: 64x32 output tile per wave (8 accumulators, A-frags reused
// against 2 B-frags -> ~43 FLOP/byte of fragment traffic, vs 16 for a
// single-tile version). Compute-bound workload (~200 GFLOP, fits in L2).
// ---------------------------------------------------------------------------

typedef __attribute__((ext_vector_type(16))) __bf16 v16bf;
typedef __attribute__((ext_vector_type(8)))  float  v8f;
typedef __attribute__((ext_vector_type(4)))  unsigned int u32x4;

__device__ __forceinline__ uint16_t f2bf(float f) {
  union { float f; uint32_t u; } x; x.f = f;
  uint32_t r = x.u + 0x7fffu + ((x.u >> 16) & 1u);
  return (uint16_t)(r >> 16);
}
__device__ __forceinline__ float bf2f(uint16_t h) {
  union { uint32_t u; float f; } x; x.u = ((uint32_t)h) << 16; return x.f;
}
__device__ __forceinline__ float wred_sum(float v) {
  for (int o = 16; o > 0; o >>= 1) v += __shfl_xor(v, o, 32);
  return v;
}
__device__ __forceinline__ float wred_max(float v) {
  for (int o = 16; o > 0; o >>= 1) v = fmaxf(v, __shfl_xor(v, o, 32));
  return v;
}
__device__ __forceinline__ float sigm(float x) { return 1.f / (1.f + __expf(-x)); }

union ABfrag { v16bf v; u32x4 q[2]; };

// ---------------------------------------------------------------------------
// Fat-tile WMMA GEMM: C[M,N] = act(alpha * A[M,K] @ W[N,K]^T + bias + madd)
// One wave computes a 64x32 output tile (4 M-subtiles x 2 N-subtiles).
// Requires M%64==0, N%32==0, K%32==0, lda/ldw multiples of 8.
// ---------------------------------------------------------------------------
__global__ __launch_bounds__(32)
void k_gemm64(const uint16_t* __restrict__ A, int lda,
              const uint16_t* __restrict__ W, int ldw,
              const float* __restrict__ bias, const float* __restrict__ madd,
              float alpha, int act,
              float* __restrict__ Cf, uint16_t* __restrict__ Cb, int ldc, int K)
{
  const int lane = threadIdx.x;
  const int r    = lane & 15;
  const int hi   = lane >> 4;
  const size_t tm = (size_t)blockIdx.x * 64;
  const size_t tn = (size_t)blockIdx.y * 32;
  // A fragment rows (16x32 bf16): lanes 0-15 -> K 0..7 / 16..23, lanes16-31 -> +8
  const uint16_t* ap0 = A + (tm +  0 + r) * (size_t)lda + (hi << 3);
  const uint16_t* ap1 = A + (tm + 16 + r) * (size_t)lda + (hi << 3);
  const uint16_t* ap2 = A + (tm + 32 + r) * (size_t)lda + (hi << 3);
  const uint16_t* ap3 = A + (tm + 48 + r) * (size_t)lda + (hi << 3);
  // B fragment (32x16 bf16): lanes 0-15 -> K 0..7 / 16..23, lanes16-31 -> +8
  const uint16_t* wp0 = W + (tn +  0 + r) * (size_t)ldw + (hi << 3);
  const uint16_t* wp1 = W + (tn + 16 + r) * (size_t)ldw + (hi << 3);
  v8f c[8];
#pragma unroll
  for (int i = 0; i < 8; ++i) c[i] = (v8f){0.f,0.f,0.f,0.f,0.f,0.f,0.f,0.f};
  for (int k0 = 0; k0 < K; k0 += 32) {
    ABfrag b0, b1, a0, a1, a2, a3;
    b0.q[0] = *(const u32x4*)(wp0 + k0);  b0.q[1] = *(const u32x4*)(wp0 + k0 + 16);
    b1.q[0] = *(const u32x4*)(wp1 + k0);  b1.q[1] = *(const u32x4*)(wp1 + k0 + 16);
    a0.q[0] = *(const u32x4*)(ap0 + k0);  a0.q[1] = *(const u32x4*)(ap0 + k0 + 16);
    a1.q[0] = *(const u32x4*)(ap1 + k0);  a1.q[1] = *(const u32x4*)(ap1 + k0 + 16);
    a2.q[0] = *(const u32x4*)(ap2 + k0);  a2.q[1] = *(const u32x4*)(ap2 + k0 + 16);
    a3.q[0] = *(const u32x4*)(ap3 + k0);  a3.q[1] = *(const u32x4*)(ap3 + k0 + 16);
    c[0] = __builtin_amdgcn_wmma_f32_16x16x32_bf16(false, a0.v, false, b0.v, (short)0, c[0], false, false);
    c[1] = __builtin_amdgcn_wmma_f32_16x16x32_bf16(false, a0.v, false, b1.v, (short)0, c[1], false, false);
    c[2] = __builtin_amdgcn_wmma_f32_16x16x32_bf16(false, a1.v, false, b0.v, (short)0, c[2], false, false);
    c[3] = __builtin_amdgcn_wmma_f32_16x16x32_bf16(false, a1.v, false, b1.v, (short)0, c[3], false, false);
    c[4] = __builtin_amdgcn_wmma_f32_16x16x32_bf16(false, a2.v, false, b0.v, (short)0, c[4], false, false);
    c[5] = __builtin_amdgcn_wmma_f32_16x16x32_bf16(false, a2.v, false, b1.v, (short)0, c[5], false, false);
    c[6] = __builtin_amdgcn_wmma_f32_16x16x32_bf16(false, a3.v, false, b0.v, (short)0, c[6], false, false);
    c[7] = __builtin_amdgcn_wmma_f32_16x16x32_bf16(false, a3.v, false, b1.v, (short)0, c[7], false, false);
  }
  float bv[2];
#pragma unroll
  for (int ni = 0; ni < 2; ++ni) {
    int col = (int)tn + ni * 16 + r;
    bv[ni] = 0.f;
    if (bias) bv[ni] += bias[col];
    if (madd) bv[ni] += madd[col];
  }
#pragma unroll
  for (int mi = 0; mi < 4; ++mi) {
#pragma unroll
    for (int ni = 0; ni < 2; ++ni) {
      const v8f cc = c[mi * 2 + ni];
      int col = (int)tn + ni * 16 + r;
#pragma unroll
      for (int v = 0; v < 8; ++v) {
        int row = (int)tm + mi * 16 + v + (hi << 3);
        float val = alpha * cc[v] + bv[ni];
        if (act) val = fmaxf(val, 0.f);
        size_t o = (size_t)row * (size_t)ldc + col;
        if (Cf) Cf[o] = val;
        if (Cb) Cb[o] = f2bf(val);
      }
    }
  }
}

// Fallback WMMA GEMM (16x16 tile per wave) for shapes not divisible by 64/32.
__global__ __launch_bounds__(32)
void k_gemm16(const uint16_t* __restrict__ A, int lda,
              const uint16_t* __restrict__ W, int ldw,
              const float* __restrict__ bias, const float* __restrict__ madd,
              float alpha, int act,
              float* __restrict__ Cf, uint16_t* __restrict__ Cb, int ldc, int K)
{
  const int lane = threadIdx.x;
  const int r    = lane & 15;
  const int hi   = lane >> 4;
  const size_t tm = (size_t)blockIdx.x * 16;
  const size_t tn = (size_t)blockIdx.y * 16;
  const uint16_t* ap = A + (tm + r) * (size_t)lda + (hi << 3);
  const uint16_t* wp = W + (tn + r) * (size_t)ldw + (hi << 3);
  v8f c = {0.f, 0.f, 0.f, 0.f, 0.f, 0.f, 0.f, 0.f};
  for (int k0 = 0; k0 < K; k0 += 32) {
    ABfrag a, b;
    a.q[0] = *(const u32x4*)(ap + k0);
    a.q[1] = *(const u32x4*)(ap + k0 + 16);
    b.q[0] = *(const u32x4*)(wp + k0);
    b.q[1] = *(const u32x4*)(wp + k0 + 16);
    c = __builtin_amdgcn_wmma_f32_16x16x32_bf16(false, a.v, false, b.v,
                                                (short)0, c, false, false);
  }
  const int col = (int)tn + r;
  float bv = 0.f;
  if (bias) bv += bias[col];
  if (madd) bv += madd[col];
#pragma unroll
  for (int v = 0; v < 8; ++v) {
    int row = (int)tm + v + (hi << 3);
    float val = alpha * c[v] + bv;
    if (act) val = fmaxf(val, 0.f);
    size_t o = (size_t)row * (size_t)ldc + col;
    if (Cf) Cf[o] = val;
    if (Cb) Cb[o] = f2bf(val);
  }
}

// f32 -> bf16 with column padding (c >= cols written as 0)
__global__ void k_cvt_pad(uint16_t* __restrict__ dst, int ldd,
                          const float* __restrict__ src, int lds,
                          int rows, int cols)
{
  int idx = blockIdx.x * blockDim.x + threadIdx.x;
  if (idx >= rows * ldd) return;
  int r = idx / ldd, c = idx % ldd;
  dst[idx] = (c < cols) ? f2bf(src[(size_t)r * lds + c]) : (uint16_t)0;
}

__global__ void k_zero16(uint16_t* __restrict__ dst, int n)
{
  int idx = blockIdx.x * blockDim.x + threadIdx.x;
  if (idx < n) dst[idx] = 0;
}

// X[t,c] = bf16(emb[H[t], c]);   512 x 128
__global__ void k_gather_emb(uint16_t* __restrict__ X, const int* __restrict__ H,
                             const float* __restrict__ emb)
{
  int idx = blockIdx.x * blockDim.x + threadIdx.x;
  if (idx >= 512 * 128) return;
  int t = idx >> 7, c = idx & 127;
  X[idx] = f2bf(emb[H[t] * 128 + c]);
}

// rbf[t, k] (16384 x 32, k>=21 zero-padded)
__global__ void k_rbf(uint16_t* __restrict__ rbf, const float* __restrict__ edge)
{
  int idx = blockIdx.x * blockDim.x + threadIdx.x;
  if (idx >= 16384 * 32) return;
  int t = idx >> 5, k = idx & 31;
  float v = 0.f;
  if (k < 21) { float d = edge[t] - 0.5f * (float)k; v = __expf(-d * d * 10.f); }
  rbf[idx] = f2bf(v);
}

// m[b,i,j,:] = [h[b,j], h[b,i], e[b,i,j]]  -> 16384 x 384 (bf16, optional f32)
__global__ void k_concat(const uint16_t* __restrict__ hb, const float* __restrict__ hf,
                         const uint16_t* __restrict__ eb, const float* __restrict__ ef,
                         uint16_t* __restrict__ db, float* __restrict__ df)
{
  int idx = blockIdx.x * blockDim.x + threadIdx.x;
  if (idx >= 16384 * 384) return;
  int t = idx / 384, c = idx % 384;
  int b = t >> 10, ij = t & 1023, i = ij >> 5, j = ij & 31;
  int s;
  const uint16_t* sb; const float* sf;
  if (c < 128)      { s = (b * 32 + j) * 128 + c;         sb = hb; sf = hf; }
  else if (c < 256) { s = (b * 32 + i) * 128 + (c - 128); sb = hb; sf = hf; }
  else              { s = t * 128 + (c - 256);            sb = eb; sf = ef; }
  db[idx] = sb[s];
  if (df) df[idx] = sf[s];
}

// GRU fusion: out = (1-z)*n + z*h (+ h if residual); gi/gh are (M, 3d) f32.
__global__ void k_gru_fuse(const float* __restrict__ gi, const float* __restrict__ gh,
                           const float* __restrict__ hprev, int ldh,
                           float* __restrict__ outf, uint16_t* __restrict__ outb, int ldo,
                           int M, int d, int residual)
{
  int idx = blockIdx.x * blockDim.x + threadIdx.x;
  if (idx >= M * d) return;
  int t = idx / d, c = idx % d;
  size_t gr = (size_t)t * (3 * d);
  float ir = gi[gr + c], iz = gi[gr + d + c], ig = gi[gr + 2 * d + c];
  float hr = gh[gr + c], hz = gh[gr + d + c], hg = gh[gr + 2 * d + c];
  float rr = sigm(ir + hr);
  float zz = sigm(iz + hz);
  float nn = tanhf(ig + rr * hg);
  float h  = hprev[(size_t)t * ldh + c];
  float val = (1.f - zz) * nn + zz * h + (residual ? h : 0.f);
  size_t o = (size_t)t * ldo + c;
  outf[o] = val;
  outb[o] = f2bf(val);
}

// V[(b,j), e', c] = sum_d x[b,j,d] * Wm[d*32+e', c];  x = h[:, j*32 .. +32)
__global__ void k_towerV(float* __restrict__ V, const float* __restrict__ x /* +j*32, ld 128 */,
                         const float* __restrict__ Wm)
{
  int idx = blockIdx.x * blockDim.x + threadIdx.x;   // 512*32*128
  if (idx >= 512 * 32 * 128) return;
  int c = idx & 127, ep = (idx >> 7) & 31, bj = idx >> 12;
  const float* xp = x + (size_t)bj * 128;
  float s = 0.f;
#pragma unroll 4
  for (int d = 0; d < 32; ++d) s += xp[d] * Wm[(d * 32 + ep) * 128 + c];
  V[idx] = s;
}

// msg[b,i,e'] = sum_j A[b,i,j] * ( V[(b,j),e',:] . e[b,i,j,:] + sum_d x_d bm[d*32+e'] )
__global__ __launch_bounds__(32)
void k_tower_msg(float* __restrict__ msgf, uint16_t* __restrict__ msgb,
                 const float* __restrict__ V, const float* __restrict__ ef,
                 const float* __restrict__ Adev,
                 const float* __restrict__ x /* h_f32 + j*32, ld 128 */,
                 const float* __restrict__ bm)
{
  int bi = blockIdx.x;           // b*32 + i
  int b = bi >> 5, i = bi & 31;
  int ep = threadIdx.x;
  float acc = 0.f;
  for (int j = 0; j < 32; ++j) {
    float a = Adev[(b * 32 + i) * 32 + j];
    if (a != 0.f) {
      const float* Vp = V + ((size_t)(b * 32 + j) * 32 + ep) * 128;
      const float* ev = ef + (size_t)(b * 1024 + i * 32 + j) * 128;
      float s = 0.f;
#pragma unroll 4
      for (int c = 0; c < 128; ++c) s += Vp[c] * ev[c];
      const float* xp = x + (size_t)(b * 32 + j) * 128;
      float sb = 0.f;
#pragma unroll 4
      for (int d = 0; d < 32; ++d) sb += xp[d] * bm[d * 32 + ep];
      acc += a * (s + sb);
    }
  }
  msgf[bi * 32 + ep] = acc;
  msgb[bi * 32 + ep] = f2bf(acc);
}

__global__ void k_maskbuf(float* __restrict__ m, const float* __restrict__ Adev)
{
  int idx = blockIdx.x * blockDim.x + threadIdx.x;
  if (idx < 16384) m[idx] = (Adev[idx] == 0.f) ? -1.0e9f : 0.f;
}

// softmax over a 1024-wide row of f32 scores -> bf16 attn
__global__ __launch_bounds__(32)
void k_softmax(const float* __restrict__ sc, uint16_t* __restrict__ attn)
{
  int t = blockIdx.x, lane = threadIdx.x;
  float v[32];
  float mx = -3.4e38f;
#pragma unroll
  for (int i = 0; i < 32; ++i) {
    v[i] = sc[(size_t)t * 1024 + lane + 32 * i];
    mx = fmaxf(mx, v[i]);
  }
  mx = wred_max(mx);
  float s = 0.f;
#pragma unroll
  for (int i = 0; i < 32; ++i) { v[i] = __expf(v[i] - mx); s += v[i]; }
  s = wred_sum(s);
  float inv = 1.f / s;
#pragma unroll
  for (int i = 0; i < 32; ++i)
    attn[(size_t)t * 1024 + lane + 32 * i] = f2bf(v[i] * inv);
}

// vt[d, s] = src[s*1152 + d]   (src pre-offset to V block of one (b,h))
__global__ void k_vt(uint16_t* __restrict__ vt, const uint16_t* __restrict__ src)
{
  int idx = blockIdx.x * blockDim.x + threadIdx.x;   // 96*1024
  if (idx >= 96 * 1024) return;
  int d = idx >> 10, s = idx & 1023;
  vt[idx] = src[(size_t)s * 1152 + d];
}

// out = LayerNorm(x + y) * g + be, D = 384; writes f32 (may alias x) and bf16.
__global__ __launch_bounds__(32)
void k_add_ln(const float* __restrict__ x, const float* __restrict__ y,
              const float* __restrict__ g, const float* __restrict__ be,
              float* __restrict__ outf, uint16_t* __restrict__ outb)
{
  int t = blockIdx.x, lane = threadIdx.x;
  float v[12];
  float s = 0.f;
#pragma unroll
  for (int i = 0; i < 12; ++i) {
    int c = lane + 32 * i;
    v[i] = x[(size_t)t * 384 + c] + y[(size_t)t * 384 + c];
    s += v[i];
  }
  s = wred_sum(s);
  float mu = s * (1.f / 384.f);
  float q = 0.f;
#pragma unroll
  for (int i = 0; i < 12; ++i) { float d = v[i] - mu; q += d * d; }
  q = wred_sum(q);
  float rstd = rsqrtf(q * (1.f / 384.f) + 1e-5f);
#pragma unroll
  for (int i = 0; i < 12; ++i) {
    int c = lane + 32 * i;
    float val = (v[i] - mu) * rstd * g[c] + be[c];
    outf[(size_t)t * 384 + c] = val;
    outb[(size_t)t * 384 + c] = f2bf(val);
  }
}

// head final linear: out[t] = x[t,:512] . W + b
__global__ __launch_bounds__(32)
void k_head_final(const uint16_t* __restrict__ x, const float* __restrict__ W,
                  const float* __restrict__ b, float* __restrict__ out)
{
  int t = blockIdx.x, lane = threadIdx.x;
  float acc = 0.f;
#pragma unroll
  for (int i = 0; i < 16; ++i) {
    int c = lane + 32 * i;
    acc += bf2f(x[(size_t)t * 512 + c]) * W[c];
  }
  acc = wred_sum(acc);
  if (lane == 0) out[t] = acc + b[0];
}

__global__ __launch_bounds__(32)
void k_masked_mean(const float* __restrict__ v, const float* __restrict__ Adev,
                   float* __restrict__ out)
{
  int b = blockIdx.x, lane = threadIdx.x;
  float num = 0.f, den = 0.f;
  for (int s = lane; s < 1024; s += 32) {
    float a = Adev[b * 1024 + s];
    num += v[b * 1024 + s] * a;
    den += a;
  }
  num = wred_sum(num);
  den = wred_sum(den);
  if (lane == 0) out[b] = num / den;
}

// ---------------------------------------------------------------------------
extern "C" void kernel_launch(void* const* d_in, const int* in_sizes, int n_in,
                              void* d_out, int out_size, void* d_ws, size_t ws_size,
                              hipStream_t stream)
{
  (void)in_sizes; (void)n_in; (void)out_size; (void)ws_size;
  const int*   H    = (const int*)d_in[0];
  const float* Adev = (const float*)d_in[1];
  const float* edge = (const float*)d_in[2];
  auto f = [&](int i) { return (const float*)d_in[i]; };

  // params flattened: top keys sorted: W_ee, W_ne, emb, head, layers, trans
  int p = 3;
  const float* W_ee = f(p++);   // (128,21)
  const float* W_ne = f(p++);   // (128,128)
  const float* emb  = f(p++);   // (4,128)
  const float *hdW[4], *hdB[4];
  for (int i = 0; i < 4; ++i) { hdW[i] = f(p++); hdB[i] = f(p++); }
  struct TowerP { const float *bh, *bi, *wh, *wi, *mW, *mB; };
  struct LayerP {
    const float *cW, *cB, *g_bh, *g_bi, *g_wh, *g_wi;
    const float *mlpW[5], *mlpB[5];
    TowerP tw[4];
  } L[3];
  for (int l = 0; l < 3; ++l) {
    L[l].cW = f(p++); L[l].cB = f(p++);                       // concat
    L[l].g_bh = f(p++); L[l].g_bi = f(p++);                   // edge_gru (bh,bi,wh,wi)
    L[l].g_wh = f(p++); L[l].g_wi = f(p++);
    for (int i = 0; i < 5; ++i) { L[l].mlpW[i] = f(p++); L[l].mlpB[i] = f(p++); }
    for (int j = 0; j < 4; ++j) {
      L[l].tw[j].bh = f(p++); L[l].tw[j].bi = f(p++);
      L[l].tw[j].wh = f(p++); L[l].tw[j].wi = f(p++);
      L[l].tw[j].mW = f(p++); L[l].tw[j].mB = f(p++);
    }
  }
  struct TransP { const float *Wo, *Wqkv, *be1, *be2, *bo, *bqkv,
                              *ff1W, *ff1b, *ff2W, *ff2b, *g1, *g2; } T[2];
  for (int t = 0; t < 2; ++t) {
    T[t].Wo = f(p++); T[t].Wqkv = f(p++); T[t].be1 = f(p++); T[t].be2 = f(p++);
    T[t].bo = f(p++); T[t].bqkv = f(p++);
    T[t].ff1W = f(p++); T[t].ff1b = f(p++); T[t].ff2W = f(p++); T[t].ff2b = f(p++);
    T[t].g1 = f(p++); T[t].g2 = f(p++);
  }

  // ---- workspace layout (bump allocator, explicit aliasing for reuse) ----
  char* ws = (char*)d_ws;
  size_t off = 0;
  auto alloc = [&](size_t b) { size_t o = off; off += (b + 255) & ~(size_t)255; return ws + o; };
  uint16_t* wbuf    = (uint16_t*)alloc(1152 * 384 * 2);       // weight bf16 scratch
  uint16_t* xemb    = (uint16_t*)alloc(512 * 128 * 2);
  float*    h_f     = (float*)   alloc(512 * 128 * 4);
  uint16_t* h_b     = (uint16_t*)alloc(512 * 128 * 2);
  float*    ht_f    = (float*)   alloc(512 * 128 * 4);
  uint16_t* ht_b    = (uint16_t*)alloc(512 * 128 * 2);
  uint16_t* rbf_b   = (uint16_t*)alloc(16384 * 32 * 2);
  float*    maskb   = (float*)   alloc(16384 * 4);
  float*    msg_f   = (float*)   alloc(512 * 32 * 4);
  uint16_t* msg_b   = (uint16_t*)alloc(512 * 32 * 2);
  float*    tgi     = (float*)   alloc(512 * 96 * 4);
  float*    tgh     = (float*)   alloc(512 * 96 * 4);
  uint16_t* ff1_b   = (uint16_t*)alloc(16384 * 32 * 2);       // K padded to 32
  float*    scores  = (float*)   alloc((size_t)1024 * 1024 * 4);
  uint16_t* attn_b  = (uint16_t*)alloc((size_t)1024 * 1024 * 2);
  uint16_t* vt_b    = (uint16_t*)alloc(96 * 1024 * 2);
  float*    headv   = (float*)   alloc(16384 * 4);
  float*    e_f     = (float*)   alloc((size_t)16384 * 128 * 4);
  uint16_t* e_b     = (uint16_t*)alloc((size_t)16384 * 128 * 2);
  uint16_t* mA      = (uint16_t*)alloc((size_t)16384 * 384 * 2);
  uint16_t* mB      = (uint16_t*)alloc((size_t)16384 * 384 * 2);
  float*    gi_f    = (float*)   alloc((size_t)16384 * 384 * 4);
  float*    gh_f    = (float*)   alloc((size_t)16384 * 384 * 4);
  float*    Vbuf    = (float*)   alloc((size_t)512 * 32 * 128 * 4);
  uint16_t* qkv_b   = (uint16_t*)alloc((size_t)16384 * 1152 * 2);
  float*    r_f     = (float*)   alloc((size_t)16384 * 384 * 4);
  uint16_t* r_b     = (uint16_t*)alloc((size_t)16384 * 384 * 2);
  // aliases (GNN scratch reused by transformer / head)
  float*    o_f     = gi_f;                          // 16384x384 f32
  uint16_t* o_b     = mA;                            // 16384x384 bf16
  float*    proj_f  = gh_f;                          // 16384x384 f32 (also ff out)
  uint16_t* b1_b    = qkv_b;                         // 16384x512 bf16
  uint16_t* b2_b    = qkv_b + (size_t)16384 * 512;   // 16384x512 bf16

  auto CVT = [&](uint16_t* dst, int ldd, const float* src, int lds, int rows, int cols) {
    int n = rows * ldd;
    k_cvt_pad<<<(n + 255) / 256, 256, 0, stream>>>(dst, ldd, src, lds, rows, cols);
  };
  auto GEMM = [&](const uint16_t* Ap, int lda, const uint16_t* Wp, int ldw,
                  const float* bias, const float* madd, float alpha, int act,
                  float* Cf, uint16_t* Cb, int ldc, int M, int N, int K) {
    if ((M % 64) == 0 && (N % 32) == 0) {
      dim3 g(M / 64, N / 32);
      k_gemm64<<<g, 32, 0, stream>>>(Ap, lda, Wp, ldw, bias, madd, alpha, act, Cf, Cb, ldc, K);
    } else {
      dim3 g(M / 16, N / 16);
      k_gemm16<<<g, 32, 0, stream>>>(Ap, lda, Wp, ldw, bias, madd, alpha, act, Cf, Cb, ldc, K);
    }
  };

  // zero the K-pad columns of the ff1 activation buffer once per call
  k_zero16<<<(16384 * 32 + 255) / 256, 256, 0, stream>>>(ff1_b, 16384 * 32);

  // ---- 1. node embedding: h = emb[H] @ W_ne^T  (512x128) ----
  k_gather_emb<<<(512 * 128 + 255) / 256, 256, 0, stream>>>(xemb, H, emb);
  CVT(wbuf, 128, W_ne, 128, 128, 128);
  GEMM(xemb, 128, wbuf, 128, nullptr, nullptr, 1.f, 0, h_f, h_b, 128, 512, 128, 128);

  // ---- 2. edge embedding: e = rbf(edge) @ W_ee^T  (16384x128, K padded 21->32) ----
  k_rbf<<<(16384 * 32 + 255) / 256, 256, 0, stream>>>(rbf_b, edge);
  CVT(wbuf, 32, W_ee, 21, 128, 21);
  GEMM(rbf_b, 32, wbuf, 32, nullptr, nullptr, 1.f, 0, e_f, e_b, 128, 16384, 128, 32);

  // ---- 3. GNN layers ----
  for (int l = 0; l < 3; ++l) {
    // edge MLP input: m = [h_j, h_i, e_ij]  (16384 x 384)
    k_concat<<<(16384 * 384 + 255) / 256, 256, 0, stream>>>(h_b, nullptr, e_b, nullptr, mA, nullptr);
    // 5-layer MLP: 384->256 (relu) ->256->256->256 (relu) ->128
    const int din[5]  = {384, 256, 256, 256, 256};
    const int dout[5] = {256, 256, 256, 256, 128};
    uint16_t* cur = mA;
    uint16_t* nxt = mB;
    for (int li = 0; li < 5; ++li) {
      CVT(wbuf, din[li], L[l].mlpW[li], din[li], dout[li], din[li]);
      GEMM(cur, din[li], wbuf, din[li], L[l].mlpB[li], nullptr, 1.f, li < 4 ? 1 : 0,
           nullptr, nxt, dout[li], 16384, dout[li], din[li]);
      uint16_t* t2 = cur; cur = nxt; nxt = t2;
    }
    uint16_t* mlp_out = cur;   // 16384 x 128 bf16
    // edge GRU: gi = mlp_out @ wi^T + bi ; gh = e @ wh^T + bh ; fuse
    CVT(wbuf, 128, L[l].g_wi, 128, 384, 128);
    GEMM(mlp_out, 128, wbuf, 128, L[l].g_bi, nullptr, 1.f, 0, gi_f, nullptr, 384, 16384, 384, 128);
    CVT(wbuf, 128, L[l].g_wh, 128, 384, 128);
    GEMM(e_b, 128, wbuf, 128, L[l].g_bh, nullptr, 1.f, 0, gh_f, nullptr, 384, 16384, 384, 128);
    k_gru_fuse<<<(16384 * 128 + 255) / 256, 256, 0, stream>>>(
        gi_f, gh_f, e_f, 128, e_f, e_b, 128, 16384, 128, 0);
    // towers
    for (int j = 0; j < 4; ++j) {
      const float* xj_f = h_f + j * 32;       // ld 128
      const uint16_t* xj_b = h_b + j * 32;    // ld 128
      k_towerV<<<(512 * 32 * 128 + 255) / 256, 256, 0, stream>>>(Vbuf, xj_f, L[l].tw[j].mW);
      k_tower_msg<<<512, 32, 0, stream>>>(msg_f, msg_b, Vbuf, e_f, Adev, xj_f, L[l].tw[j].mB);
      CVT(wbuf, 32, L[l].tw[j].wi, 32, 96, 32);
      GEMM(msg_b, 32, wbuf, 32, L[l].tw[j].bi, nullptr, 1.f, 0, tgi, nullptr, 96, 512, 96, 32);
      CVT(wbuf, 32, L[l].tw[j].wh, 32, 96, 32);
      GEMM(xj_b, 128, wbuf, 32, L[l].tw[j].bh, nullptr, 1.f, 0, tgh, nullptr, 96, 512, 96, 32);
      k_gru_fuse<<<(512 * 32 + 255) / 256, 256, 0, stream>>>(
          tgi, tgh, xj_f, 128, ht_f + j * 32, ht_b + j * 32, 128, 512, 32, 1);
    }
    // concat linear: h = ht @ Wc^T + bc
    CVT(wbuf, 128, L[l].cW, 128, 128, 128);
    GEMM(ht_b, 128, wbuf, 128, L[l].cB, nullptr, 1.f, 0, h_f, h_b, 128, 512, 128, 128);
  }

  // ---- 4. transformer tokens: r[b, i*32+j] = [h_j, h_i, e_ij]  (16384 x 384) ----
  k_concat<<<(16384 * 384 + 255) / 256, 256, 0, stream>>>(h_b, h_f, e_b, e_f, r_b, r_f);
  k_maskbuf<<<(16384 + 255) / 256, 256, 0, stream>>>(maskb, Adev);

  const float inv_sqrt_hd = 0.10206207262f;   // 1/sqrt(96)
  for (int t = 0; t < 2; ++t) {
    // QKV projection
    CVT(wbuf, 384, T[t].Wqkv, 384, 1152, 384);
    GEMM(r_b, 384, wbuf, 384, T[t].bqkv, nullptr, 1.f, 0, nullptr, qkv_b, 1152, 16384, 1152, 384);
    // per-(batch, head) attention
    for (int b = 0; b < 16; ++b) {
      for (int hh = 0; hh < 4; ++hh) {
        const uint16_t* qp = qkv_b + (size_t)(b * 1024) * 1152 + hh * 96;
        const uint16_t* kp = qp + 384;
        const uint16_t* vp = qp + 768;
        // scores = (Q K^T)/sqrt(hd) + mask
        GEMM(qp, 1152, kp, 1152, nullptr, maskb + b * 1024, inv_sqrt_hd, 0,
             scores, nullptr, 1024, 1024, 1024, 96);
        k_softmax<<<1024, 32, 0, stream>>>(scores, attn_b);
        k_vt<<<(96 * 1024 + 255) / 256, 256, 0, stream>>>(vt_b, vp);
        // O_head = attn @ V  -> into o at column offset hh*96
        GEMM(attn_b, 1024, vt_b, 1024, nullptr, nullptr, 1.f, 0,
             o_f + (size_t)(b * 1024) * 384 + hh * 96,
             o_b + (size_t)(b * 1024) * 384 + hh * 96, 384, 1024, 96, 1024);
      }
    }
    // output projection + residual layernorm
    CVT(wbuf, 384, T[t].Wo, 384, 384, 384);
    GEMM(o_b, 384, wbuf, 384, T[t].bo, nullptr, 1.f, 0, proj_f, nullptr, 384, 16384, 384, 384);
    k_add_ln<<<16384, 32, 0, stream>>>(r_f, proj_f, T[t].g1, T[t].be1, r_f, r_b);
    // feed-forward 384 -> 16 (relu) -> 384, K padded 16->32
    CVT(wbuf, 384, T[t].ff1W, 384, 16, 384);
    GEMM(r_b, 384, wbuf, 384, T[t].ff1b, nullptr, 1.f, 1, nullptr, ff1_b, 32, 16384, 16, 384);
    CVT(wbuf, 32, T[t].ff2W, 16, 384, 16);
    GEMM(ff1_b, 32, wbuf, 32, T[t].ff2b, nullptr, 1.f, 0, proj_f, nullptr, 384, 16384, 384, 32);
    k_add_ln<<<16384, 32, 0, stream>>>(r_f, proj_f, T[t].g2, T[t].be2, r_f, r_b);
  }

  // ---- 5. head: 384->512->512->512 (relu) -> 1, then masked mean ----
  CVT(wbuf, 384, hdW[0], 384, 512, 384);
  GEMM(r_b, 384, wbuf, 384, hdB[0], nullptr, 1.f, 1, nullptr, b1_b, 512, 16384, 512, 384);
  CVT(wbuf, 512, hdW[1], 512, 512, 512);
  GEMM(b1_b, 512, wbuf, 512, hdB[1], nullptr, 1.f, 1, nullptr, b2_b, 512, 16384, 512, 512);
  CVT(wbuf, 512, hdW[2], 512, 512, 512);
  GEMM(b2_b, 512, wbuf, 512, hdB[2], nullptr, 1.f, 1, nullptr, b1_b, 512, 16384, 512, 512);
  k_head_final<<<16384, 32, 0, stream>>>(b1_b, hdW[3], hdB[3], headv);
  k_masked_mean<<<16, 32, 0, stream>>>(headv, Adev, (float*)d_out);
}
